// SelfAttentionModel_39848706573222
// MI455X (gfx1250) — matrix operand order repaired and checked by reading, not compiled, not run on hardware
//
#include <hip/hip_runtime.h>
#include <hip/hip_bf16.h>
#include <math.h>

// ---------------------------------------------------------------------------
// SelfAttention for B=2, S=2048, E=1024, H=16, D=64 on gfx1250 (CDNA5).
// All matmul FLOPs go through v_wmma_f32_16x16x32_bf16 (wave32 WMMA).
//
// v2 changes vs v1:
//  - projection GEMM: wave tile 32x64 (8 wmma / 32-K chunk, 12 b128 loads
//    -> 1.5 loads/wmma, 8 independent accumulator chains)
//  - attention: 32 query rows per wave (two 16-row A tiles share every
//    K/V fragment -> 0.67 loads/wmma, 2x wmma density per iteration)
//  - vectorized fp32->bf16 conversion (float4 -> bf16x4)
// ---------------------------------------------------------------------------

typedef __bf16 bf16;
typedef __attribute__((ext_vector_type(16))) __bf16 v16bf;
typedef __attribute__((ext_vector_type(8)))  __bf16 v8bf;
typedef __attribute__((ext_vector_type(4)))  __bf16 v4bf;
typedef __attribute__((ext_vector_type(8)))  float  v8f;

#define B_   2
#define H_   16
#define S_   2048
#define D_   64
#define E_   1024

static __device__ __forceinline__ v8f wmma_bf16(v16bf a, v16bf b, v8f c) {
  // (neg_a, A, neg_b, B, c_mod, C, reuse_a, reuse_b)
  return __builtin_amdgcn_wmma_f32_16x16x32_bf16(false, a, false, b, (short)0, c,
                                                 false, false);
}

// A-matrix fragment (16x32 bf16, MxK). Lane L: row M = L%16 (caller bakes the
// row into `row`), holds K = 8g+{0..7} (VGPR0-3) and 16+8g+{0..7} (VGPR4-7).
static __device__ __forceinline__ v16bf load_frag_a(const bf16* __restrict__ row, int g) {
  v8bf lo = *reinterpret_cast<const v8bf*>(row + 8 * g);
  v8bf hi = *reinterpret_cast<const v8bf*>(row + 16 + 8 * g);
  return __builtin_shufflevector(lo, hi, 0, 1, 2, 3, 4, 5, 6, 7,
                                 8, 9, 10, 11, 12, 13, 14, 15);
}

// B-matrix fragment (32x16 bf16, KxN). Lane L: col N = L%16 (caller bakes the
// column's K-run base into `col`), holds K = 8g+{0..7} and 16+8g+{0..7}.
static __device__ __forceinline__ v16bf load_frag_b(const bf16* __restrict__ col, int g) {
  v8bf lo = *reinterpret_cast<const v8bf*>(col + 8 * g);
  v8bf hi = *reinterpret_cast<const v8bf*>(col + 16 + 8 * g);
  return __builtin_shufflevector(lo, hi, 0, 1, 2, 3, 4, 5, 6, 7,
                                 8, 9, 10, 11, 12, 13, 14, 15);
}

// ---------------------------------------------------------------------------
// Kernel 1: fp32 -> bf16 conversion, 4 elems/thread (one-time, lives in L2)
// ---------------------------------------------------------------------------
__global__ void sa_cvt_f32_bf16(const float* __restrict__ src, bf16* __restrict__ dst,
                                int n4) {
  int i = blockIdx.x * blockDim.x + threadIdx.x;
  if (i < n4) {
    const float4 f = reinterpret_cast<const float4*>(src)[i];
    v4bf o;
    o[0] = (bf16)f.x; o[1] = (bf16)f.y; o[2] = (bf16)f.z; o[3] = (bf16)f.w;
    reinterpret_cast<v4bf*>(dst)[i] = o;
  }
}

// ---------------------------------------------------------------------------
// Kernel 2: projection GEMM  Y = X @ W^T + b, scattered into head layout.
// X: [4096 x 1024] bf16, W: [1024 x 1024] bf16 (row n of W = column n of W^T,
// contiguous along k). Wave -> 32x64 tile (8 wmma/k-chunk); block = 8 waves
// (2 along M, 4 along N) -> 64x256 block tile.
// transposeOut==0: Y as [B,H,S,D]; transposeOut==1: Y as [B,H,D,S] (for V).
// ---------------------------------------------------------------------------
__global__ void sa_proj_kernel(const bf16* __restrict__ X, const bf16* __restrict__ W,
                               const float* __restrict__ bias, bf16* __restrict__ Y,
                               int transposeOut) {
  constexpr int K = E_;
  const int wave = threadIdx.x >> 5;
  const int lane = threadIdx.x & 31;
  const int g = lane >> 4;
  const int ln = lane & 15;

  const int m0 = blockIdx.x * 64 + (wave & 1) * 32;
  const int n0 = blockIdx.y * 256 + (wave >> 1) * 64;

  v8f acc[2][4] = {};

  const bf16* arow0 = X + (size_t)(m0 + ln) * K;
  const bf16* arow1 = X + (size_t)(m0 + 16 + ln) * K;
  const bf16* brow[4];
#pragma unroll
  for (int j = 0; j < 4; ++j) brow[j] = W + (size_t)(n0 + j * 16 + ln) * K;

  for (int kk = 0; kk < K; kk += 32) {
    const v16bf a0 = load_frag_a(arow0 + kk, g);
    const v16bf a1 = load_frag_a(arow1 + kk, g);
#pragma unroll
    for (int j = 0; j < 4; ++j) {
      const v16bf b = load_frag_b(brow[j] + kk, g);
      acc[0][j] = wmma_bf16(a0, b, acc[0][j]);
      acc[1][j] = wmma_bf16(a1, b, acc[1][j]);
    }
  }

  // C/D layout: lane L, VGPR r -> (M = r + 8*(L/16), N = L%16)
#pragma unroll
  for (int mi = 0; mi < 2; ++mi) {
#pragma unroll
    for (int j = 0; j < 4; ++j) {
      const int mt = m0 + mi * 16;
      const int n = n0 + j * 16 + ln;
      const float bval = bias[n];
      const int h = n >> 6, d = n & 63;
#pragma unroll
      for (int r = 0; r < 8; ++r) {
        const int m = mt + r + 8 * g;
        const int bb = m >> 11, s = m & 2047;
        const float v = acc[mi][j][r] + bval;
        size_t idx;
        if (transposeOut)
          idx = ((size_t)(bb * H_ + h) * D_ + d) * S_ + s;   // [B,H,D,S]
        else
          idx = ((size_t)(bb * H_ + h) * S_ + s) * D_ + d;   // [B,H,S,D]
        Y[idx] = (bf16)v;
      }
    }
  }
}

// ---------------------------------------------------------------------------
// Kernel 3: flash attention. One wave = 32 query rows (two 16-row tiles that
// share all K/V fragments); 8 waves/block -> 256 query rows per block.
// Online softmax over 32-key chunks; P goes C-layout -> A-layout via LDS.
// ---------------------------------------------------------------------------
__global__ void sa_attn_kernel(const bf16* __restrict__ qh, const bf16* __restrict__ kh,
                               const bf16* __restrict__ vT, const int* __restrict__ mask,
                               float* __restrict__ out) {
  const int bb = blockIdx.z;
  const int h = blockIdx.y;
  const int wave = threadIdx.x >> 5;
  const int lane = threadIdx.x & 31;
  const int g = lane >> 4;
  const int ln = lane & 15;
  const int q0 = blockIdx.x * 256 + wave * 32;

  const bf16* qbase = qh + ((size_t)(bb * H_ + h) * S_) * D_;
  const bf16* kbase = kh + ((size_t)(bb * H_ + h) * S_) * D_;
  const bf16* vbase = vT + ((size_t)(bb * H_ + h) * D_) * S_;
  const int* mrow = mask + (size_t)bb * S_;

  // Q fragments: aq[mi][c] -> rows q0+16*mi+ln, d chunk c (0..31 / 32..63)
  v16bf aq[2][2];
#pragma unroll
  for (int mi = 0; mi < 2; ++mi) {
    const bf16* qrow = qbase + (size_t)(q0 + mi * 16 + ln) * D_;
    aq[mi][0] = load_frag_a(qrow, g);
    aq[mi][1] = load_frag_a(qrow + 32, g);
  }

  v8f acc[2][4] = {};
  float mrun[2][8], lrun[2][8];
#pragma unroll
  for (int mi = 0; mi < 2; ++mi)
#pragma unroll
    for (int r = 0; r < 8; ++r) { mrun[mi][r] = -INFINITY; lrun[mi][r] = 0.0f; }

  __shared__ __align__(16) __bf16 pSh[8][2][16][32];  // per-wave two 16x32 P tiles

  const float scale = 0.03125f;  // 1/sqrt(1024)

  for (int kb = 0; kb < S_; kb += 32) {
    if (kb + 32 < S_) {
      __builtin_prefetch(kbase + (size_t)(kb + 32 + ln) * D_, 0, 1);
      __builtin_prefetch(vbase + (size_t)ln * S_ + kb + 32, 0, 1);
    }

    // ---- K fragments shared by both query tiles ----
    v16bf kf[2][2];   // [key subtile t][d chunk]
    bool masked[2];
#pragma unroll
    for (int t = 0; t < 2; ++t) {
      const int sk = kb + t * 16 + ln;          // this lane's key column
      const bf16* krow = kbase + (size_t)sk * D_;
      kf[t][0] = load_frag_b(krow, g);          // d = 8g..8g+7, 16+8g..16+8g+7
      kf[t][1] = load_frag_b(krow + 32, g);     // d = 32+8g.., 48+8g..
      masked[t] = (mrow[sk] == 0);
    }

    // ---- scores: 2 m-tiles x 2 key subtiles, K-dim = 64 (2 wmma each) ----
    v8f sc[2][2];
#pragma unroll
    for (int mi = 0; mi < 2; ++mi)
#pragma unroll
      for (int t = 0; t < 2; ++t) {
        v8f s = {};
        s = wmma_bf16(aq[mi][0], kf[t][0], s);
        s = wmma_bf16(aq[mi][1], kf[t][1], s);
#pragma unroll
        for (int r = 0; r < 8; ++r) s[r] = masked[t] ? -INFINITY : s[r] * scale;
        sc[mi][t] = s;
      }

    // ---- online softmax (rows live in 16-lane half-wave groups) ----
    v8f pf[2][2];
#pragma unroll
    for (int mi = 0; mi < 2; ++mi) {
#pragma unroll
      for (int r = 0; r < 8; ++r) {
        float rm = fmaxf(sc[mi][0][r], sc[mi][1][r]);
        rm = fmaxf(rm, __shfl_xor(rm, 1));
        rm = fmaxf(rm, __shfl_xor(rm, 2));
        rm = fmaxf(rm, __shfl_xor(rm, 4));
        rm = fmaxf(rm, __shfl_xor(rm, 8));
        const float mnew = fmaxf(mrun[mi][r], rm);
        const float corr = __expf(mrun[mi][r] - mnew);
        const float p0 = __expf(sc[mi][0][r] - mnew);
        const float p1 = __expf(sc[mi][1][r] - mnew);
        float rs = p0 + p1;
        rs += __shfl_xor(rs, 1);
        rs += __shfl_xor(rs, 2);
        rs += __shfl_xor(rs, 4);
        rs += __shfl_xor(rs, 8);
        lrun[mi][r] = lrun[mi][r] * corr + rs;
        mrun[mi][r] = mnew;
        pf[mi][0][r] = p0;
        pf[mi][1][r] = p1;
        acc[mi][0][r] *= corr;
        acc[mi][1][r] *= corr;
        acc[mi][2][r] *= corr;
        acc[mi][3][r] *= corr;
      }
    }

    // ---- relayout P: C-layout -> A-fragment via LDS bounce ----
    __syncthreads();
#pragma unroll
    for (int mi = 0; mi < 2; ++mi)
#pragma unroll
      for (int t = 0; t < 2; ++t)
#pragma unroll
        for (int r = 0; r < 8; ++r)
          pSh[wave][mi][r + 8 * g][t * 16 + ln] = (bf16)pf[mi][t][r];
    __syncthreads();
    v16bf ap[2];
#pragma unroll
    for (int mi = 0; mi < 2; ++mi)
      ap[mi] = load_frag_a(&pSh[wave][mi][ln][0], g);  // A: 16 rows x 32 keys

    // ---- PV: out[32x64] += P[32x32] x V[32x64]; V frags shared by m-tiles --
#pragma unroll
    for (int nt = 0; nt < 4; ++nt) {
      const bf16* vrow = vbase + (size_t)(nt * 16 + ln) * S_ + kb;  // col d, keys kb..kb+31
      const v16bf bvf = load_frag_b(vrow, g);
      acc[0][nt] = wmma_bf16(ap[0], bvf, acc[0][nt]);
      acc[1][nt] = wmma_bf16(ap[1], bvf, acc[1][nt]);
    }
  }

  // ---- epilogue: divide by l, store fp32 [B,H,S,D] ----
  float* obase = out + ((size_t)(bb * H_ + h) * S_) * D_;
#pragma unroll
  for (int mi = 0; mi < 2; ++mi) {
#pragma unroll
    for (int nt = 0; nt < 4; ++nt) {
#pragma unroll
      for (int r = 0; r < 8; ++r) {
        const int m = q0 + mi * 16 + r + 8 * g;
        const int d = nt * 16 + ln;
        obase[(size_t)m * D_ + d] = acc[mi][nt][r] / lrun[mi][r];
      }
    }
  }
}

// ---------------------------------------------------------------------------
// Host launch
// ---------------------------------------------------------------------------
extern "C" void kernel_launch(void* const* d_in, const int* in_sizes, int n_in,
                              void* d_out, int out_size, void* d_ws, size_t ws_size,
                              hipStream_t stream) {
  const float* query = (const float*)d_in[0];
  const float* key   = (const float*)d_in[1];
  const float* value = (const float*)d_in[2];
  const int*   amask = (const int*)d_in[3];
  const float* Wq = (const float*)d_in[4];
  const float* bq = (const float*)d_in[5];
  const float* Wk = (const float*)d_in[6];
  const float* bk = (const float*)d_in[7];
  const float* Wv = (const float*)d_in[8];
  const float* bv = (const float*)d_in[9];
  float* out = (float*)d_out;

  const size_t M4 = (size_t)4096 * 1024;   // activation elems
  const size_t M1 = (size_t)1024 * 1024;   // weight elems

  bf16* ws = (bf16*)d_ws;
  bf16* XQ = ws;
  bf16* XK = XQ + M4;
  bf16* XV = XK + M4;
  bf16* WQb = XV + M4;
  bf16* WKb = WQb + M1;
  bf16* WVb = WKb + M1;
  bf16* QH = WVb + M1;        // [B,H,S,D]
  bf16* KH = QH + M4;         // [B,H,S,D]
  bf16* VT = KH + M4;         // [B,H,D,S]

  // 1) fp32 -> bf16 (vectorized x4)
  const unsigned g4 = (unsigned)((M4 / 4 + 255) / 256);
  const unsigned g1 = (unsigned)((M1 / 4 + 255) / 256);
  sa_cvt_f32_bf16<<<g4, 256, 0, stream>>>(query, XQ, (int)(M4 / 4));
  sa_cvt_f32_bf16<<<g4, 256, 0, stream>>>(key,   XK, (int)(M4 / 4));
  sa_cvt_f32_bf16<<<g4, 256, 0, stream>>>(value, XV, (int)(M4 / 4));
  sa_cvt_f32_bf16<<<g1, 256, 0, stream>>>(Wq, WQb, (int)(M1 / 4));
  sa_cvt_f32_bf16<<<g1, 256, 0, stream>>>(Wk, WKb, (int)(M1 / 4));
  sa_cvt_f32_bf16<<<g1, 256, 0, stream>>>(Wv, WVb, (int)(M1 / 4));

  // 2) projections (4096x1024 @ 1024x1024 each), block tile 64x256
  dim3 pg(4096 / 64, 1024 / 256);
  sa_proj_kernel<<<pg, 256, 0, stream>>>(XQ, WQb, bq, QH, 0);
  sa_proj_kernel<<<pg, 256, 0, stream>>>(XK, WKb, bk, KH, 0);
  sa_proj_kernel<<<pg, 256, 0, stream>>>(XV, WVb, bv, VT, 1);

  // 3) attention: 256 query rows per block
  dim3 ag(S_ / 256, H_, B_);
  sa_attn_kernel<<<ag, 256, 0, stream>>>(QH, KH, VT, amask, out);
}
